// Attention_21131239096383
// MI455X (gfx1250) — matrix operand order repaired and checked by reading, not compiled, not run on hardware
//
#include <hip/hip_runtime.h>
#include <hip/hip_bf16.h>

typedef __bf16 bf16_t;
typedef __attribute__((ext_vector_type(16))) __bf16 bf16x16;
typedef __attribute__((ext_vector_type(8)))  float  f32x8;
typedef int v4i __attribute__((vector_size(16)));   // matches async builtin's pointee

constexpr int SEQ    = 2048;
constexpr int DIM    = 4096;
constexpr int NHEADS = 32;
constexpr int NKV    = 8;
constexpr int HD     = 128;
constexpr int KVDIM  = NKV * HD;   // 1024

// ---- CDNA5 async global->LDS path (probe via __has_builtin; fallback = reg copy)
#if defined(__has_builtin)
#  if __has_builtin(__builtin_amdgcn_global_load_async_to_lds_b128) && \
      __has_builtin(__builtin_amdgcn_s_wait_asynccnt)
#    define HAVE_ASYNC_LDS 1
#  else
#    define HAVE_ASYNC_LDS 0
#  endif
#else
#  define HAVE_ASYNC_LDS 0
#endif

#if HAVE_ASYNC_LDS
#  define WAIT_ASYNC(n) __builtin_amdgcn_s_wait_asynccnt(n)
#else
#  define WAIT_ASYNC(n)
#endif

__device__ __forceinline__ void copy16_to_lds(bf16_t* lds, const bf16_t* g) {
#if HAVE_ASYNC_LDS
  __builtin_amdgcn_global_load_async_to_lds_b128(
      (__attribute__((address_space(1))) v4i*)g,
      (__attribute__((address_space(3))) v4i*)lds, 0, 0);
#else
  *(uint4*)lds = *(const uint4*)g;
#endif
}

union FragU { bf16x16 v; uint4 u[2]; };
union H8    { uint4 u; __bf16 b[8]; };

__device__ __forceinline__ f32x8 wmma_bf16(bf16x16 a, bf16x16 b, f32x8 c) {
  // D = A(16x32) * B(32x16) + C, f32 accum
  return __builtin_amdgcn_wmma_f32_16x16x32_bf16(false, a, false, b, (short)0, c,
                                                 false, false);
}

// A fragment (16x32 bf16). p = this lane's row pointer (row = lane&15), K contiguous.
__device__ __forceinline__ bf16x16 load_a_frag(const bf16_t* p, int hl) {
  FragU f;
  f.u[0] = *(const uint4*)(p + hl * 8);
  f.u[1] = *(const uint4*)(p + 16 + hl * 8);
  return f.v;
}

// B fragment (32x16 bf16). base -> [n0][k0] of an [n][k] tile, `stride` elems per n.
__device__ __forceinline__ bf16x16 load_b_frag(const bf16_t* base, int stride, int lane) {
  const bf16_t* p = base + (size_t)(lane & 15) * stride + (lane >> 4) * 8;
  FragU f;
  f.u[0] = *(const uint4*)(p);
  f.u[1] = *(const uint4*)(p + 16);
  return f.v;
}

// ---------------------------------------------------------------- conversions
__global__ void cvt_kernel(const float* __restrict__ in, bf16_t* __restrict__ out,
                           long n) {
  long i = (long)blockIdx.x * blockDim.x + threadIdx.x;
  if (i < n) out[i] = (bf16_t)in[i];
}

// --------------------------------------------------------------------- GEMM
// C[M,N] = A[M,K] * B[N,K]^T ; bf16 in, f32 accum, f32 or bf16 out.
// Block 128x128, 8 waves, wave tile 32x64; double-buffered LDS, async fills.
__global__ __launch_bounds__(256) void gemm_bt_kernel(
    const bf16_t* __restrict__ A, const bf16_t* __restrict__ B,
    float* __restrict__ Cf, bf16_t* __restrict__ Cb, int M, int N, int K) {
  __shared__ __align__(16) bf16_t As[2][128 * 32];
  __shared__ __align__(16) bf16_t Bs[2][128 * 32];
  const int tid  = threadIdx.x;
  const int wave = tid >> 5, lane = tid & 31;
  const int hl = lane >> 4, ln = lane & 15;
  const int wm = (wave & 3) * 32;     // 0,32,64,96
  const int wn = (wave >> 2) * 64;    // 0,64
  const size_t bm = (size_t)blockIdx.y * 128;
  const size_t bn = (size_t)blockIdx.x * 128;

  f32x8 acc[2][4] = {};

  // stage one 128x32 A tile + 128x32 B tile (4 x b128 per thread -> ASYNCcnt 4/wave)
  auto issue_tile = [&](int buf, int k0) {
#pragma unroll
    for (int i = 0; i < 2; ++i) {
      int chunk = tid + i * 256;              // 512 x 16B chunks per matrix
      int r  = chunk >> 2;
      int c8 = (chunk & 3) * 8;
      copy16_to_lds(&As[buf][r * 32 + c8], &A[(bm + r) * K + k0 + c8]);
      copy16_to_lds(&Bs[buf][r * 32 + c8], &B[(bn + r) * K + k0 + c8]);
    }
  };

  const int ntiles = K >> 5;
  issue_tile(0, 0);

  for (int t = 0; t < ntiles; ++t) {
    const int b = t & 1;
    if (t + 1 < ntiles) {
      issue_tile(1 - b, (t + 1) * 32);        // overlap next fill with this compute
      WAIT_ASYNC(4);                          // tile t landed (in-order), t+1 in flight
      if (t + 2 < ntiles) {                   // keep L2 warm two tiles ahead
        __builtin_prefetch(&A[(bm + (tid >> 1)) * K + (t + 2) * 32], 0, 1);
        __builtin_prefetch(&B[(bn + (tid >> 1)) * K + (t + 2) * 32], 0, 1);
      }
    } else {
      WAIT_ASYNC(0);
    }
    __syncthreads();                          // tile t visible to all waves

    bf16x16 bfr[4];
#pragma unroll
    for (int ni = 0; ni < 4; ++ni)
      bfr[ni] = load_b_frag(&Bs[b][(wn + ni * 16) * 32], 32, lane);
#pragma unroll
    for (int mi = 0; mi < 2; ++mi) {
      bf16x16 afr = load_a_frag(&As[b][(wm + mi * 16 + ln) * 32], hl);
#pragma unroll
      for (int ni = 0; ni < 4; ++ni)
        acc[mi][ni] = wmma_bf16(afr, bfr[ni], acc[mi][ni]);
    }
    __syncthreads();                          // done reading buf b before t+1 refills it
  }

  // Epilogue: C/D layout — lane col = ln, VGPR r holds row r + 8*hl.
#pragma unroll
  for (int mi = 0; mi < 2; ++mi)
#pragma unroll
    for (int ni = 0; ni < 4; ++ni)
#pragma unroll
      for (int r = 0; r < 8; ++r) {
        size_t row = bm + wm + mi * 16 + r + 8 * hl;
        size_t col = bn + wn + ni * 16 + ln;
        if (Cb) Cb[row * N + col] = (bf16_t)acc[mi][ni][r];
        else    Cf[row * N + col] = acc[mi][ni][r];
      }
}

// --------------------------------------------------------------------- RoPE
__global__ void rope_kernel(const bf16_t* __restrict__ in, bf16_t* __restrict__ out,
                            const float* __restrict__ cosp,
                            const float* __restrict__ sinp, int nh) {
  int idx = blockIdx.x * blockDim.x + threadIdx.x;
  int total = SEQ * nh * (HD / 2);
  if (idx >= total) return;
  int i = idx & 63;
  int t = idx >> 6;
  int h = t % nh;
  int s = t / nh;
  size_t base = (size_t)s * nh * HD + (size_t)h * HD;
  float x1 = (float)in[base + 2 * i];
  float x2 = (float)in[base + 2 * i + 1];
  float c  = cosp[s * 64 + i];
  float sn = sinp[s * 64 + i];
  out[base + i]      = (bf16_t)(x1 * c - x2 * sn);
  out[base + 64 + i] = (bf16_t)(x2 * c + x1 * sn);
}

// ------------------------------------------------------------ Flash attention
// Block = (64 q-rows, 1 head), 4 waves x 16 q-rows. Online softmax, causal.
__global__ __launch_bounds__(128) void flash_attn_kernel(
    const bf16_t* __restrict__ Q, const bf16_t* __restrict__ Kc,
    const bf16_t* __restrict__ Vc, bf16_t* __restrict__ ctx) {
  __shared__ __align__(16) bf16_t Ks[32 * 128];      // [key][d]
  __shared__ __align__(16) bf16_t Vt[128 * 32];      // [d][key] (transposed)
  __shared__ __align__(16) bf16_t Ps[4][16 * 32];    // per-wave P tile [q][key]

  const int tid  = threadIdx.x, wave = tid >> 5, lane = tid & 31;
  const int hl = lane >> 4, ln = lane & 15;
  const int head = blockIdx.y, kvh = head >> 2;      // GQA: 4 q-heads per kv-head
  const int qblk = blockIdx.x * 64;
  const int qw   = qblk + wave * 16;

  const bf16_t* qp = Q + (size_t)(qw + ln) * DIM + head * HD;
  bf16x16 qf[4];
#pragma unroll
  for (int c = 0; c < 4; ++c) qf[c] = load_a_frag(qp + c * 32, hl);

  f32x8 o[8] = {};
  float mrow[8], lrow[8];
#pragma unroll
  for (int r = 0; r < 8; ++r) { mrow[r] = -3.0e38f; lrow[r] = 0.0f; }
  const float scale = 0.08838834764831845f;          // 1/sqrt(128)

  const int kend = qblk + 64;                        // causal bound (uniform per block)
  for (int kc = 0; kc < kend; kc += 32) {
    __syncthreads();
#pragma unroll
    for (int i = 0; i < 4; ++i) {                    // stage K async; V via reg transpose
      int chunk = tid + i * 128;
      int row = chunk >> 4;
      int c8  = (chunk & 15) * 8;
      size_t g = (size_t)(kc + row) * KVDIM + (size_t)kvh * HD + c8;
      copy16_to_lds(&Ks[row * 128 + c8], &Kc[g]);
      H8 vv; vv.u = *(const uint4*)&Vc[g];
#pragma unroll
      for (int j = 0; j < 8; ++j) Vt[(c8 + j) * 32 + row] = vv.b[j];
    }
    WAIT_ASYNC(0);
    __syncthreads();

    // S = Q K^T : two 16x16 score tiles (keys kc..+15, kc+16..+31)
    f32x8 s0 = {}, s1 = {};
#pragma unroll
    for (int c = 0; c < 4; ++c) {
      bf16x16 b0 = load_b_frag(&Ks[c * 32], 128, lane);
      bf16x16 b1 = load_b_frag(&Ks[16 * 128 + c * 32], 128, lane);
      s0 = wmma_bf16(qf[c], b0, s0);
      s1 = wmma_bf16(qf[c], b1, s1);
    }

    // Online softmax; row r+8*hl lives across the 16 lanes of this half-wave.
#pragma unroll
    for (int r = 0; r < 8; ++r) {
      const int q  = qw + r + 8 * hl;
      const int k0 = kc + ln, k1 = k0 + 16;
      float v0 = (k0 <= q) ? s0[r] * scale : -3.0e38f;
      float v1 = (k1 <= q) ? s1[r] * scale : -3.0e38f;
      float cm = fmaxf(v0, v1);
#pragma unroll
      for (int m = 8; m >= 1; m >>= 1) cm = fmaxf(cm, __shfl_xor(cm, m, 32));
      const float nm = fmaxf(mrow[r], cm);
      const float p0 = __expf(v0 - nm);
      const float p1 = __expf(v1 - nm);
      float rs = p0 + p1;
#pragma unroll
      for (int m = 8; m >= 1; m >>= 1) rs += __shfl_xor(rs, m, 32);
      const float corr = __expf(mrow[r] - nm);
      lrow[r] = lrow[r] * corr + rs;
      mrow[r] = nm;
#pragma unroll
      for (int n = 0; n < 8; ++n) o[n][r] *= corr;
      Ps[wave][(r + 8 * hl) * 32 + ln]      = (bf16_t)p0;   // C->A relayout via LDS
      Ps[wave][(r + 8 * hl) * 32 + 16 + ln] = (bf16_t)p1;
    }

    // O += P V  (P: 16x32 A-frag, V: 32x16 B-frags per 16-wide d tile)
    bf16x16 af = load_a_frag(&Ps[wave][ln * 32], hl);
#pragma unroll
    for (int n = 0; n < 8; ++n) {
      bf16x16 bv = load_b_frag(&Vt[n * 16 * 32], 32, lane);
      o[n] = wmma_bf16(af, bv, o[n]);
    }
  }

#pragma unroll
  for (int r = 0; r < 8; ++r) {
    const float inv = 1.0f / lrow[r];
    const size_t q = (size_t)(qw + r + 8 * hl);
#pragma unroll
    for (int n = 0; n < 8; ++n)
      ctx[q * DIM + head * HD + n * 16 + ln] = (bf16_t)(o[n][r] * inv);
  }
}

// ------------------------------------------------------------------- launcher
extern "C" void kernel_launch(void* const* d_in, const int* in_sizes, int n_in,
                              void* d_out, int out_size, void* d_ws, size_t ws_size,
                              hipStream_t stream) {
  (void)in_sizes; (void)n_in; (void)out_size; (void)ws_size;
  const float* x    = (const float*)d_in[0];
  /* d_in[1] = attention_mask (causal tril) — implicit in the kernel */
  const float* cosp = (const float*)d_in[2];
  const float* sinp = (const float*)d_in[3];
  const float* wq   = (const float*)d_in[4];
  const float* wk   = (const float*)d_in[5];
  const float* wv   = (const float*)d_in[6];
  const float* wo   = (const float*)d_in[7];
  float* out = (float*)d_out;

  bf16_t* wsb = (bf16_t*)d_ws;
  size_t off = 0;
  auto carve = [&](size_t n) { bf16_t* p = wsb + off; off += n; return p; };
  bf16_t* xb   = carve((size_t)SEQ * DIM);
  bf16_t* wqb  = carve((size_t)DIM * DIM);
  bf16_t* wkb  = carve((size_t)KVDIM * DIM);
  bf16_t* wvb  = carve((size_t)KVDIM * DIM);
  bf16_t* wob  = carve((size_t)DIM * DIM);
  bf16_t* qraw = carve((size_t)SEQ * DIM);
  bf16_t* kraw = carve((size_t)SEQ * KVDIM);
  bf16_t* qb   = carve((size_t)SEQ * DIM);
  bf16_t* kb   = carve((size_t)SEQ * KVDIM);
  bf16_t* vb   = carve((size_t)SEQ * KVDIM);
  bf16_t* ctx  = carve((size_t)SEQ * DIM);

  auto cdiv = [](long a, long b) { return (unsigned)((a + b - 1) / b); };

  cvt_kernel<<<cdiv((long)SEQ * DIM, 256), 256, 0, stream>>>(x,  xb,  (long)SEQ * DIM);
  cvt_kernel<<<cdiv((long)DIM * DIM, 256), 256, 0, stream>>>(wq, wqb, (long)DIM * DIM);
  cvt_kernel<<<cdiv((long)KVDIM * DIM, 256), 256, 0, stream>>>(wk, wkb, (long)KVDIM * DIM);
  cvt_kernel<<<cdiv((long)KVDIM * DIM, 256), 256, 0, stream>>>(wv, wvb, (long)KVDIM * DIM);
  cvt_kernel<<<cdiv((long)DIM * DIM, 256), 256, 0, stream>>>(wo, wob, (long)DIM * DIM);

  gemm_bt_kernel<<<dim3(DIM / 128, SEQ / 128), 256, 0, stream>>>(
      xb, wqb, nullptr, qraw, SEQ, DIM, DIM);
  gemm_bt_kernel<<<dim3(KVDIM / 128, SEQ / 128), 256, 0, stream>>>(
      xb, wkb, nullptr, kraw, SEQ, KVDIM, DIM);
  gemm_bt_kernel<<<dim3(KVDIM / 128, SEQ / 128), 256, 0, stream>>>(
      xb, wvb, nullptr, vb, SEQ, KVDIM, DIM);

  rope_kernel<<<cdiv((long)SEQ * NHEADS * 64, 256), 256, 0, stream>>>(
      qraw, qb, cosp, sinp, NHEADS);
  rope_kernel<<<cdiv((long)SEQ * NKV * 64, 256), 256, 0, stream>>>(
      kraw, kb, cosp, sinp, NKV);

  flash_attn_kernel<<<dim3(SEQ / 64, NHEADS), 128, 0, stream>>>(qb, kb, vb, ctx);

  gemm_bt_kernel<<<dim3(DIM / 128, SEQ / 128), 256, 0, stream>>>(
      ctx, wob, out, nullptr, SEQ, DIM, DIM);
}
